// NGP_50886772523404
// MI455X (gfx1250) — matrix operand order repaired and checked by reading, not compiled, not run on hardware
//
#include <hip/hip_runtime.h>
#include <hip/hip_bf16.h>

typedef _Float16 v8h  __attribute__((ext_vector_type(8)));
typedef _Float16 v16h __attribute__((ext_vector_type(16)));
typedef float    v8f  __attribute__((ext_vector_type(8)));

#define TSZ   (1u << 19)
#define TMASK (TSZ - 1u)

// ---- WMMA fragment helpers (wave32; contiguous K order shared by A and B, not the ISA table order) ----
// A (16x32 f16): lane m = L&15, h = L>>4; halfs j -> K = h*16 + j
static __device__ inline v16h afrag(const _Float16* row, int chunk, int h) {
    const v8h lo = *(const v8h*)(row + chunk * 32 + h * 16);
    const v8h hi = *(const v8h*)(row + chunk * 32 + h * 16 + 8);
    return __builtin_shufflevector(lo, hi, 0,1,2,3,4,5,6,7,8,9,10,11,12,13,14,15);
}
// B (32x16 f16): lane n = L&15, h = L>>4; halfs j -> K = h*16 + j.  WT is [N][Kdim] (K contiguous).
static __device__ inline v16h bfrag(const _Float16* WT, int Kdim, int n, int chunk, int h) {
    const _Float16* base = WT + n * Kdim + chunk * 32 + h * 16;
    const v8h lo = *(const v8h*)base;
    const v8h hi = *(const v8h*)(base + 8);
    return __builtin_shufflevector(lo, hi, 0,1,2,3,4,5,6,7,8,9,10,11,12,13,14,15);
}
static __device__ inline v8f wmma16(v16h a, v16h b, v8f c) {
    return __builtin_amdgcn_wmma_f32_16x16x32_f16(false, a, false, b, (short)0, c, false, false);
}

__global__ __launch_bounds__(256)
void ngp_fused(const float* __restrict__ x, const float* __restrict__ rdir,
               const float* __restrict__ tables,
               const float* __restrict__ dW1, const float* __restrict__ db1,
               const float* __restrict__ dW2, const float* __restrict__ db2,
               const float* __restrict__ cW1, const float* __restrict__ cb1,
               const float* __restrict__ cW2, const float* __restrict__ cb2,
               const float* __restrict__ cW3, const float* __restrict__ cb3,
               float* __restrict__ outColor, float* __restrict__ outSigma, int P)
{
    // ---- LDS: transposed f16 weights [N][K], biases, per-wave activation buffers ----
    __shared__ alignas(16) _Float16 sWT1[64 * 32];   // dW1^T (K=32, N=64)
    __shared__ alignas(16) _Float16 sWT2[16 * 64];   // dW2^T (K=64, N=16)
    __shared__ alignas(16) _Float16 sWT3[64 * 64];   // cW1^T (K=43 pad 64, N=64)
    __shared__ alignas(16) _Float16 sWT4[64 * 64];   // cW2^T
    __shared__ alignas(16) _Float16 sWT5[16 * 64];   // cW3^T (N=3 pad 16)
    __shared__ float sBias[224];                     // db1|db2|cb1|cb2|cb3(pad)
    __shared__ float sLev[16];
    __shared__ alignas(16) _Float16 sBufA[8][16 * 64];
    __shared__ alignas(16) _Float16 sBufB[8][16 * 64];
    __shared__ float sMask[8][16];

    const int tid = threadIdx.x;
    // ---- cooperative weight staging (transpose + f16 convert + zero-pad) ----
    for (int i = tid; i < 64 * 32; i += 256) { int n = i >> 5, k = i & 31; sWT1[i] = (_Float16)dW1[k * 64 + n]; }
    for (int i = tid; i < 16 * 64; i += 256) { int n = i >> 6, k = i & 63; sWT2[i] = (_Float16)dW2[k * 16 + n]; }
    for (int i = tid; i < 64 * 64; i += 256) { int n = i >> 6, k = i & 63; sWT3[i] = (k < 43) ? (_Float16)cW1[k * 64 + n] : (_Float16)0.f; }
    for (int i = tid; i < 64 * 64; i += 256) { int n = i >> 6, k = i & 63; sWT4[i] = (_Float16)cW2[k * 64 + n]; }
    for (int i = tid; i < 16 * 64; i += 256) { int n = i >> 6, k = i & 63; sWT5[i] = (n < 3) ? (_Float16)cW3[k * 3 + n] : (_Float16)0.f; }
    for (int i = tid; i < 224; i += 256) {
        float b = 0.f;
        if (i < 64) b = db1[i];
        else if (i < 80) b = db2[i - 64];
        else if (i < 144) b = cb1[i - 80];
        else if (i < 208) b = cb2[i - 144];
        else if (i < 211) b = cb3[i - 208];
        sBias[i] = b;
    }
    if (tid == 0) {  // np.geomspace(16, 2048, 16).astype(int)
        sLev[0]=16.f;  sLev[1]=22.f;  sLev[2]=30.f;   sLev[3]=42.f;
        sLev[4]=58.f;  sLev[5]=80.f;  sLev[6]=111.f;  sLev[7]=153.f;
        sLev[8]=212.f; sLev[9]=294.f; sLev[10]=406.f; sLev[11]=561.f;
        sLev[12]=776.f;sLev[13]=1072.f;sLev[14]=1482.f;sLev[15]=2048.f;
    }
    __syncthreads();

    const int wave = tid >> 5;
    const int lane = tid & 31;
    const int m = lane & 15;      // row (M) for A, column (N) within tile for B/D
    const int h = lane >> 4;      // lane-half
    const int tileBase = blockIdx.x * 128 + wave * 16;
    const int pt  = tileBase + m;
    const int ptc = (pt < P) ? pt : (P - 1);

    _Float16* bufA = &sBufA[wave][0];
    _Float16* bufB = &sBufB[wave][0];
    _Float16* rowA = bufA + m * 64;

    // ---- position + mask ----
    const float xf0 = x[ptc * 3 + 0] + 0.5f;
    const float xf1 = x[ptc * 3 + 1] + 0.5f;
    const float xf2 = x[ptc * 3 + 2] + 0.5f;
    const bool mask = (xf0 > 0.f) & (xf0 < 1.f) & (xf1 > 0.f) & (xf1 < 1.f) & (xf2 > 0.f) & (xf2 < 1.f);
    if (h == 0) sMask[wave][m] = mask ? 1.f : 0.f;

    // ---- multiresolution hash encode: lane handles 8 levels of its point ----
    const float2* tbl2 = (const float2*)tables;
    #pragma unroll
    for (int t = 0; t < 8; ++t) {
        const int lv = h * 8 + t;
        const float s = sLev[lv];
        const float px = xf0 * s, py = xf1 * s, pz = xf2 * s;
        const float flx = floorf(px), fly = floorf(py), flz = floorf(pz);
        const float frx = px - flx, fry = py - fly, frz = pz - flz;
        const uint32_t hx0 = (uint32_t)(int)flx;
        const uint32_t hy0 = (uint32_t)(int)fly * 2654435761u;
        const uint32_t hz0 = (uint32_t)(int)flz * 805459861u;
        const uint32_t hx1 = (uint32_t)(int)ceilf(px);
        const uint32_t hy1 = (uint32_t)(int)ceilf(py) * 2654435761u;
        const uint32_t hz1 = (uint32_t)(int)ceilf(pz) * 805459861u;
        float f0 = 0.f, f1 = 0.f;
        #pragma unroll
        for (int c = 0; c < 8; ++c) {
            const uint32_t hsh = (((c & 1) ? hx1 : hx0) ^ ((c & 2) ? hy1 : hy0) ^ ((c & 4) ? hz1 : hz0)) & TMASK;
            const float2 cf = tbl2[(size_t)lv * TSZ + hsh];
            const float w = ((c & 1) ? frx : 1.f - frx) * ((c & 2) ? fry : 1.f - fry) * ((c & 4) ? frz : 1.f - frz);
            f0 = fmaf(w, cf.x, f0);
            f1 = fmaf(w, cf.y, f1);
        }
        rowA[2 * lv]     = (_Float16)f0;
        rowA[2 * lv + 1] = (_Float16)f1;
    }

    // ---- density layer 1: feats(16x32) @ dW1 -> relu hidden(16x64) in bufB ----
    {
        const v16h a0 = afrag(rowA, 0, h);
        #pragma unroll
        for (int nt = 0; nt < 4; ++nt) {
            const v16h b = bfrag(sWT1, 32, nt * 16 + m, 0, h);
            v8f acc = {};
            acc = wmma16(a0, b, acc);
            const float bv = sBias[nt * 16 + m];
            #pragma unroll
            for (int r = 0; r < 8; ++r) {
                float v = acc[r] + bv;
                v = v > 0.f ? v : 0.f;
                bufB[(r + h * 8) * 64 + nt * 16 + m] = (_Float16)v;
            }
        }
    }

    // ---- density layer 2: hidden(16x64) @ dW2 -> log_sigma(16x16); sigma out; feed bufA[:, 0:16] ----
    {
        const _Float16* rowB = bufB + m * 64;
        const v16h a0 = afrag(rowB, 0, h);
        const v16h a1 = afrag(rowB, 1, h);
        const v16h b0 = bfrag(sWT2, 64, m, 0, h);
        const v16h b1 = bfrag(sWT2, 64, m, 1, h);
        v8f acc = {};
        acc = wmma16(a0, b0, acc);
        acc = wmma16(a1, b1, acc);
        const float bv = sBias[64 + m];
        float ls[8];
        #pragma unroll
        for (int r = 0; r < 8; ++r) {
            const float v = acc[r] + bv;
            ls[r] = v;
            bufA[(r + h * 8) * 64 + m] = (_Float16)v;
        }
        if (m == 0) {  // lanes 0 and 16 hold column 0 (log_sigma[:,0])
            #pragma unroll
            for (int r = 0; r < 8; ++r) {
                const int mr = r + h * 8;
                const float sg = (sMask[wave][mr] != 0.f) ? __expf(ls[r]) : 0.f;
                if (tileBase + mr < P) outSigma[tileBase + mr] = sg;
            }
        }
    }

    // ---- direction positional encoding -> bufA[:, 16:43], zero pad to 64 ----
    {
        const float r0 = rdir[ptc * 3 + 0];
        const float r1 = rdir[ptc * 3 + 1];
        const float r2 = rdir[ptc * 3 + 2];
        const float tp = 6.2831853071795864f;
        if (h == 0) {
            rowA[16] = (_Float16)r0; rowA[17] = (_Float16)r1; rowA[18] = (_Float16)r2;
            #pragma unroll
            for (int l = 0; l < 4; ++l) {
                const float f = tp * (float)(1 << l);
                rowA[19 + l] = (_Float16)__sinf(r0 * f);
                rowA[23 + l] = (_Float16)__cosf(r0 * f);
                rowA[27 + l] = (_Float16)__sinf(r1 * f);
                rowA[31 + l] = (_Float16)__cosf(r1 * f);
            }
        } else {
            #pragma unroll
            for (int l = 0; l < 4; ++l) {
                const float f = tp * (float)(1 << l);
                rowA[35 + l] = (_Float16)__sinf(r2 * f);
                rowA[39 + l] = (_Float16)__cosf(r2 * f);
            }
            #pragma unroll
            for (int k2 = 43; k2 < 64; ++k2) rowA[k2] = (_Float16)0.f;
        }
    }

    // ---- color layer 1: (16x64) @ cW1pad -> relu -> bufB ; color layer 2 -> relu -> bufA ----
    #pragma unroll
    for (int layer = 0; layer < 2; ++layer) {
        const _Float16* src = (layer == 0) ? bufA : bufB;
        _Float16* dst       = (layer == 0) ? bufB : bufA;
        const _Float16* WT  = (layer == 0) ? sWT3 : sWT4;
        const float* bias   = (layer == 0) ? (sBias + 80) : (sBias + 144);
        const _Float16* row = src + m * 64;
        const v16h a0 = afrag(row, 0, h);
        const v16h a1 = afrag(row, 1, h);
        #pragma unroll
        for (int nt = 0; nt < 4; ++nt) {
            const v16h b0 = bfrag(WT, 64, nt * 16 + m, 0, h);
            const v16h b1 = bfrag(WT, 64, nt * 16 + m, 1, h);
            v8f acc = {};
            acc = wmma16(a0, b0, acc);
            acc = wmma16(a1, b1, acc);
            const float bv = bias[nt * 16 + m];
            #pragma unroll
            for (int r = 0; r < 8; ++r) {
                float v = acc[r] + bv;
                v = v > 0.f ? v : 0.f;
                dst[(r + h * 8) * 64 + nt * 16 + m] = (_Float16)v;
            }
        }
    }

    // ---- color layer 3: (16x64) @ cW3pad -> sigmoid, mask, store RGB ----
    {
        const _Float16* row = bufA + m * 64;
        const v16h a0 = afrag(row, 0, h);
        const v16h a1 = afrag(row, 1, h);
        const v16h b0 = bfrag(sWT5, 64, m, 0, h);
        const v16h b1 = bfrag(sWT5, 64, m, 1, h);
        v8f acc = {};
        acc = wmma16(a0, b0, acc);
        acc = wmma16(a1, b1, acc);
        if (m < 3) {
            const float bv = sBias[208 + m];
            #pragma unroll
            for (int r = 0; r < 8; ++r) {
                const int mr = r + h * 8;
                const float v = acc[r] + bv;
                float sg = 1.f / (1.f + __expf(-v));
                sg = (sMask[wave][mr] != 0.f) ? sg : 0.f;
                if (tileBase + mr < P) outColor[(size_t)(tileBase + mr) * 3 + m] = sg;
            }
        }
    }
}

extern "C" void kernel_launch(void* const* d_in, const int* in_sizes, int n_in,
                              void* d_out, int out_size, void* d_ws, size_t ws_size,
                              hipStream_t stream) {
    const float* x      = (const float*)d_in[0];
    const float* rdir   = (const float*)d_in[1];
    const float* tables = (const float*)d_in[2];
    const float* dW1 = (const float*)d_in[3];  const float* db1 = (const float*)d_in[4];
    const float* dW2 = (const float*)d_in[5];  const float* db2 = (const float*)d_in[6];
    const float* cW1 = (const float*)d_in[7];  const float* cb1 = (const float*)d_in[8];
    const float* cW2 = (const float*)d_in[9];  const float* cb2 = (const float*)d_in[10];
    const float* cW3 = (const float*)d_in[11]; const float* cb3 = (const float*)d_in[12];

    const int P = in_sizes[0] / 3;               // 393216 points
    float* outColor = (float*)d_out;             // (P,3)
    float* outSigma = (float*)d_out + (size_t)P * 3;  // (P,1)

    const int grid = (P + 127) / 128;            // 128 points per 256-thread block (8 waves x 16)
    ngp_fused<<<grid, 256, 0, stream>>>(x, rdir, tables,
                                        dW1, db1, dW2, db2,
                                        cW1, cb1, cW2, cb2, cW3, cb3,
                                        outColor, outSigma, P);
}
